// FeatureWiseAffine_22771916603486
// MI455X (gfx1250) — compile-verified, matrix operand order checked by reading, not run on hardware
//
#include <hip/hip_runtime.h>
#include <hip/hip_bf16.h>

typedef __attribute__((ext_vector_type(16))) _Float16 v16h;
typedef __attribute__((ext_vector_type(4)))  _Float16 v4h;
typedef __attribute__((ext_vector_type(8)))  float    v8f;

#define DIM   64
#define CHUNK 128
#define NTOK  16384   // H*W = 128*128
#define LN_EPS 1e-5f
#define FP    72      // f16 row pad: 144B, 16B-aligned rows -> b128 DS loads
#define GP    264     // gelu buffer pad: 528B, 16B-aligned

// ---- unified WMMA fragment loader (A layout; B uses it on [N][Kpad] tiles) --
// lane<16 -> row=lane, elems 0..7 = K0..7, 8..15 = K16..23
// lane>=16 -> same row, elems 0..7 = K8..15, 8..15 = K24..31
__device__ __forceinline__ v16h frag16(const _Float16* S, int ld, int row0, int k0) {
  const int lane = threadIdx.x & 31;
  const int hf = lane >> 4, m = lane & 15;
  const _Float16* r = S + (row0 + m) * ld + k0 + hf * 8;
  v16h a;
#pragma unroll
  for (int e = 0; e < 8; ++e) a[e] = r[e];
  r += 16;
#pragma unroll
  for (int e = 0; e < 8; ++e) a[e + 8] = r[e];
  return a;
}

// One 16-row slab per wave (8 waves cover M=128). NT n-tiles, KT k-tiles of 32.
template <int NT, int KT, typename FA, typename FB, typename EPI>
__device__ __forceinline__ void run_gemm(FA fa, FB fb, EPI epi) {
  const int row0 = (int)(threadIdx.x >> 5) * 16;
#pragma unroll
  for (int nt = 0; nt < NT; ++nt) {
    v8f c = {};
#pragma unroll
    for (int kt = 0; kt < KT; ++kt) {
      v16h a = fa(row0, kt * 32);
      v16h b = fb(kt * 32, nt * 16);
      c = __builtin_amdgcn_wmma_f32_16x16x32_f16(false, a, false, b, (short)0, c,
                                                 false, false);
    }
    epi(c, row0, nt * 16);
  }
}

// stage [K][N] row-major f32 weight -> LDS [N][kpad] f16 (transposed), coalesced
__device__ __forceinline__ void stage_w(const float* __restrict__ W, int K, int N,
                                        int kpad, _Float16* dst) {
  for (int i = threadIdx.x; i < K * N; i += 256) {
    const int k = i / N, n = i % N;   // contiguous global reads over n
    dst[n * kpad + k] = (_Float16)W[i];
  }
}
// stage [N][K] row-major f32 (conv_w is [out,in]) -> LDS [N][kpad] f16
__device__ __forceinline__ void stage_wt(const float* __restrict__ W, int N, int K,
                                         int kpad, _Float16* dst) {
  for (int i = threadIdx.x; i < N * K; i += 256) {
    const int n = i / K, k = i % K;   // contiguous global reads over k
    dst[n * kpad + k] = (_Float16)W[i];
  }
}

// ---------------- FiLM path: pooled(4x4) + 2-layer MLP -> gamma/beta ---------
__global__ __launch_bounds__(256) void film_pool_kernel(const float* __restrict__ te,
                                                        float* __restrict__ pooled) {
  const int wave = threadIdx.x >> 5, lane = threadIdx.x & 31;
  const int id = blockIdx.x * 8 + wave;         // 0..8191 = B*1024
  const int b = id >> 10, rem = id & 1023;
  const int c = rem >> 4, py = (rem >> 2) & 3, px = rem & 3;
  const float* p = te + ((size_t)b * DIM + c) * NTOK + (py * 32 + lane) * 128 + px * 32;
  float s = 0.f;
#pragma unroll
  for (int xx = 0; xx < 32; ++xx) s += p[xx];
#pragma unroll
  for (int off = 16; off > 0; off >>= 1) s += __shfl_xor(s, off, 32);
  if (lane == 0) pooled[id] = s * (1.f / 1024.f);
}

__global__ __launch_bounds__(128) void film_mlp_kernel(
    const float* __restrict__ pooled, const float* __restrict__ m1w,
    const float* __restrict__ m1b, const float* __restrict__ m2w,
    const float* __restrict__ m2b, float* __restrict__ gb) {
  __shared__ float tl[1024];
  __shared__ float hm[128];
  const int b = blockIdx.x, tid = threadIdx.x;
  for (int i = tid; i < 1024; i += 128) tl[i] = pooled[b * 1024 + i];
  __syncthreads();
  float s = m1b[tid];
  for (int i = 0; i < 1024; ++i) s += tl[i] * m1w[i * 128 + tid];
  hm[tid] = (s > 0.f) ? s : 0.01f * s;     // leaky_relu(0.01)
  __syncthreads();
  float g = m2b[tid];
#pragma unroll 8
  for (int j = 0; j < 128; ++j) g += hm[j] * m2w[j * 128 + tid];
  gb[b * 128 + tid] = g;
}

// ---------------- fused TIAM block: one workgroup per (batch, 128-tok chunk) -
__global__ __launch_bounds__(256) void tiam_fused_kernel(
    const float* __restrict__ x, const float* __restrict__ te,
    const float* __restrict__ qw, const float* __restrict__ qb,
    const float* __restrict__ kw, const float* __restrict__ kb,
    const float* __restrict__ vw, const float* __restrict__ vb,
    const float* __restrict__ ow, const float* __restrict__ ob,
    const float* __restrict__ ln1w, const float* __restrict__ ln1b,
    const float* __restrict__ ln2w, const float* __restrict__ ln2b,
    const float* __restrict__ fc1w, const float* __restrict__ fc1b,
    const float* __restrict__ fc2w, const float* __restrict__ fc2b,
    const float* __restrict__ convw, const float* __restrict__ convb,
    const float* __restrict__ gbv, float* __restrict__ out) {
  __shared__ float    s_x[CHUNK][66];          // x chunk (f32, residual)
  __shared__ float    s_p[CHUNK][66];          // prior -> h -> final out
  __shared__ _Float16 s_xh[CHUNK][FP];         // x chunk (f16, K/V GEMM A)
  __shared__ _Float16 F0[CHUNK][FP];           // LN1 -> attn-out -> LN2 -> h(f16)
  __shared__ _Float16 U[CHUNK * GP];           // union: Q/K/V (f16) OR gelu(fc1)
  __shared__ _Float16 s_w[256 * FP];           // staged weights [N][kpad] f16
  _Float16 (*Fq)[FP] = (_Float16(*)[FP])(U);
  _Float16 (*Fk)[FP] = (_Float16(*)[FP])(U + CHUNK * FP);
  _Float16 (*Fv)[FP] = (_Float16(*)[FP])(U + 2 * CHUNK * FP);
  _Float16 (*G)[GP]  = (_Float16(*)[GP])(U);   // [128][256] gelu output

  const int tid = threadIdx.x;
  const int b = blockIdx.x >> 7;
  const int n0 = (blockIdx.x & 127) * CHUNK;
  const int lane = tid & 31, lN = lane & 15, lH = lane >> 4;

  // --- load x & prior chunk, [B,C,N] -> [token][channel], coalesced on N ---
  const size_t gbase = (size_t)b * DIM * NTOK + n0;
  for (int i = tid; i < CHUNK * DIM; i += 256) {
    const int t = i & 127, c = i >> 7;
    const float xv = x[gbase + (size_t)c * NTOK + t];
    s_x[t][c] = xv;
    s_xh[t][c] = (_Float16)xv;
    s_p[t][c] = te[gbase + (size_t)c * NTOK + t];
  }
  __syncthreads();

  // --- stage q/k/v weights (transposed f16) + LN1(prior) -> F0 ---
  stage_w(qw, DIM, DIM, FP, s_w);
  stage_w(kw, DIM, DIM, FP, s_w + DIM * FP);
  stage_w(vw, DIM, DIM, FP, s_w + 2 * DIM * FP);
  if (tid < CHUNK) {
    float mu = 0.f;
#pragma unroll
    for (int c = 0; c < DIM; ++c) mu += s_p[tid][c];
    mu *= (1.f / DIM);
    float var = 0.f;
#pragma unroll
    for (int c = 0; c < DIM; ++c) { float d = s_p[tid][c] - mu; var += d * d; }
    const float rs = rsqrtf(var * (1.f / DIM) + LN_EPS);
#pragma unroll
    for (int c = 0; c < DIM; ++c)
      F0[tid][c] = (_Float16)((s_p[tid][c] - mu) * rs * ln1w[c] + ln1b[c]);
  }
  __syncthreads();

  // --- Q/K/V GEMMs (WMMA f16, f32 accum), B fragments from LDS ---
  run_gemm<4, 2>(
      [&](int r0, int k0) { return frag16(&F0[0][0], FP, r0, k0); },
      [&](int k0, int nn) { return frag16(s_w, FP, nn, k0); },
      [&](v8f c, int r0, int nn) {
#pragma unroll
        for (int r = 0; r < 8; ++r)
          Fq[r0 + 8 * lH + r][nn + lN] = (_Float16)(c[r] + qb[nn + lN]);
      });
  run_gemm<4, 2>(
      [&](int r0, int k0) { return frag16(&s_xh[0][0], FP, r0, k0); },
      [&](int k0, int nn) { return frag16(s_w + DIM * FP, FP, nn, k0); },
      [&](v8f c, int r0, int nn) {
#pragma unroll
        for (int r = 0; r < 8; ++r)
          Fk[r0 + 8 * lH + r][nn + lN] = (_Float16)(c[r] + kb[nn + lN]);
      });
  run_gemm<4, 2>(
      [&](int r0, int k0) { return frag16(&s_xh[0][0], FP, r0, k0); },
      [&](int k0, int nn) { return frag16(s_w + 2 * DIM * FP, FP, nn, k0); },
      [&](v8f c, int r0, int nn) {
#pragma unroll
        for (int r = 0; r < 8; ++r)
          Fv[r0 + 8 * lH + r][nn + lN] = (_Float16)(c[r] + vb[nn + lN]);
      });
  __syncthreads();

  // --- stage o_w + block attention (head_dim=4, flash-style online softmax) --
  stage_w(ow, DIM, DIM, FP, s_w);
  for (int p = tid; p < 16 * CHUNK; p += 256) {     // (head, q-row) pairs
    const int h = p >> 7, q = p & 127, hc = h * 4;
    const v4h qv = *(const v4h*)&Fq[q][hc];
    const float q0 = 0.5f * (float)qv[0], q1 = 0.5f * (float)qv[1];
    const float q2 = 0.5f * (float)qv[2], q3 = 0.5f * (float)qv[3];
    float m = -1e30f, l = 0.f, a0 = 0.f, a1 = 0.f, a2 = 0.f, a3 = 0.f;
#pragma unroll 4
    for (int k = 0; k < CHUNK; ++k) {
      const v4h kv = *(const v4h*)&Fk[k][hc];       // one b64 DS load
      const v4h vv = *(const v4h*)&Fv[k][hc];       // one b64 DS load
      const float s = q0 * (float)kv[0] + q1 * (float)kv[1] +
                      q2 * (float)kv[2] + q3 * (float)kv[3];
      const float nm = fmaxf(m, s);
      const float sc = __expf(m - nm);
      const float w  = __expf(s - nm);
      l  = l  * sc + w;
      a0 = a0 * sc + w * (float)vv[0];
      a1 = a1 * sc + w * (float)vv[1];
      a2 = a2 * sc + w * (float)vv[2];
      a3 = a3 * sc + w * (float)vv[3];
      m = nm;
    }
    const float inv = 1.f / l;
    F0[q][hc + 0] = (_Float16)(a0 * inv);
    F0[q][hc + 1] = (_Float16)(a1 * inv);
    F0[q][hc + 2] = (_Float16)(a2 * inv);
    F0[q][hc + 3] = (_Float16)(a3 * inv);
  }
  __syncthreads();

  // --- O-proj + bias + residual(prior) -> s_p = h ---
  run_gemm<4, 2>(
      [&](int r0, int k0) { return frag16(&F0[0][0], FP, r0, k0); },
      [&](int k0, int nn) { return frag16(s_w, FP, nn, k0); },
      [&](v8f c, int r0, int nn) {
#pragma unroll
        for (int r = 0; r < 8; ++r) {
          const int row = r0 + 8 * lH + r, n = nn + lN;
          s_p[row][n] = c[r] + ob[n] + s_p[row][n];
        }
      });
  __syncthreads();

  // --- stage fc1 weights + LN2(h) -> F0 ---
  stage_w(fc1w, DIM, 4 * DIM, FP, s_w);            // [256][72] = full buffer
  if (tid < CHUNK) {
    float mu = 0.f;
#pragma unroll
    for (int c = 0; c < DIM; ++c) mu += s_p[tid][c];
    mu *= (1.f / DIM);
    float var = 0.f;
#pragma unroll
    for (int c = 0; c < DIM; ++c) { float d = s_p[tid][c] - mu; var += d * d; }
    const float rs = rsqrtf(var * (1.f / DIM) + LN_EPS);
#pragma unroll
    for (int c = 0; c < DIM; ++c)
      F0[tid][c] = (_Float16)((s_p[tid][c] - mu) * rs * ln2w[c] + ln2b[c]);
  }
  __syncthreads();

  // --- fc1 (64->256) + exact GELU -> G (f16) ---
  run_gemm<16, 2>(
      [&](int r0, int k0) { return frag16(&F0[0][0], FP, r0, k0); },
      [&](int k0, int nn) { return frag16(s_w, FP, nn, k0); },
      [&](v8f c, int r0, int nn) {
#pragma unroll
        for (int r = 0; r < 8; ++r) {
          const int row = r0 + 8 * lH + r, n = nn + lN;
          const float v = c[r] + fc1b[n];
          G[row][n] = (_Float16)(0.5f * v * (1.f + erff(v * 0.70710678118f)));
        }
      });
  __syncthreads();

  // --- stage fc2 weights -> [64][264] ---
  stage_w(fc2w, 4 * DIM, DIM, GP, s_w);
  __syncthreads();

  // --- fc2 (256->64) + bias + residual(h) -> F0 (final h, f16) ---
  run_gemm<4, 8>(
      [&](int r0, int k0) { return frag16(&G[0][0], GP, r0, k0); },
      [&](int k0, int nn) { return frag16(s_w, GP, nn, k0); },
      [&](v8f c, int r0, int nn) {
#pragma unroll
        for (int r = 0; r < 8; ++r) {
          const int row = r0 + 8 * lH + r, n = nn + lN;
          F0[row][n] = (_Float16)(c[r] + fc2b[n] + s_p[row][n]);
        }
      });
  __syncthreads();

  // --- stage conv weights (already [out,in] -> straight copy) ---
  stage_wt(convw, DIM, DIM, FP, s_w);
  __syncthreads();

  // --- 1x1 conv + bias + x residual + FiLM -> s_p ---
  run_gemm<4, 2>(
      [&](int r0, int k0) { return frag16(&F0[0][0], FP, r0, k0); },
      [&](int k0, int nn) { return frag16(s_w, FP, nn, k0); },
      [&](v8f c, int r0, int nn) {
#pragma unroll
        for (int r = 0; r < 8; ++r) {
          const int row = r0 + 8 * lH + r, n = nn + lN;
          const float o  = c[r] + convb[n] + s_x[row][n];
          const float gm = gbv[b * 128 + n];
          const float bt = gbv[b * 128 + DIM + n];
          s_p[row][n] = (1.f + gm) * o + bt;
        }
      });
  __syncthreads();

  // --- coalesced store back to [B,C,H,W] ---
  for (int i = tid; i < CHUNK * DIM; i += 256) {
    const int t = i & 127, c = i >> 7;
    out[gbase + (size_t)c * NTOK + t] = s_p[t][c];
  }
}

extern "C" void kernel_launch(void* const* d_in, const int* in_sizes, int n_in,
                              void* d_out, int out_size, void* d_ws, size_t ws_size,
                              hipStream_t stream) {
  const float* x     = (const float*)d_in[0];
  const float* te    = (const float*)d_in[1];
  const float* qw    = (const float*)d_in[2];
  const float* qb    = (const float*)d_in[3];
  const float* kw    = (const float*)d_in[4];
  const float* kb    = (const float*)d_in[5];
  const float* vw    = (const float*)d_in[6];
  const float* vb    = (const float*)d_in[7];
  const float* ow    = (const float*)d_in[8];
  const float* ob    = (const float*)d_in[9];
  const float* ln1w  = (const float*)d_in[10];
  const float* ln1b  = (const float*)d_in[11];
  const float* ln2w  = (const float*)d_in[12];
  const float* ln2b  = (const float*)d_in[13];
  const float* fc1w  = (const float*)d_in[14];
  const float* fc1b  = (const float*)d_in[15];
  const float* fc2w  = (const float*)d_in[16];
  const float* fc2b  = (const float*)d_in[17];
  const float* convw = (const float*)d_in[18];
  const float* convb = (const float*)d_in[19];
  const float* m1w   = (const float*)d_in[20];
  const float* m1b   = (const float*)d_in[21];
  const float* m2w   = (const float*)d_in[22];
  const float* m2b   = (const float*)d_in[23];

  float* pooled = (float*)d_ws;           // 8 * 1024 floats
  float* gb     = pooled + 8 * 1024;      // 8 * 128 floats

  film_pool_kernel<<<1024, 256, 0, stream>>>(te, pooled);
  film_mlp_kernel<<<8, 128, 0, stream>>>(pooled, m1w, m1b, m2w, m2b, gb);
  tiam_fused_kernel<<<8 * 128, 256, 0, stream>>>(
      x, te, qw, qb, kw, kb, vw, vb, ow, ob, ln1w, ln1b, ln2w, ln2b,
      fc1w, fc1b, fc2w, fc2b, convw, convb, gb, (float*)d_out);
}
